// GATNet_3478923510078
// MI455X (gfx1250) — compile-verified, matrix operand order checked by reading, not run on hardware
//
#include <hip/hip_runtime.h>

typedef __attribute__((ext_vector_type(16))) _Float16 v16h;
typedef __attribute__((ext_vector_type(8)))  float    v8f;

#define N_NODES 20480
#define B_GR    512
#define NPG     40
#define E_EDGES 81920
#define E_TOT   (E_EDGES + N_NODES)   // self loops appended
#define FXD     78
#define HEADS   10
#define H1C     (HEADS * FXD)         // 780
#define OUTC    128
#define SEQ     1000
#define EMB     128
#define NF      32
#define KW      8
#define CONV_T  (SEQ - KW + 1)        // 993
#define SLOPE   0.2f

// ---------------------------------------------------------------------------
// WMMA fragment loader, per CDNA5 ISA 7.12.2 16-bit layouts (wave32).
// A (16x32, MxK): lane L holds row M=L&15; K-halves: lanes<16 -> K 0..7 &
// 16..23 in reg halves 0..7 / 8..15, lanes>=16 -> K 8..15 & 24..31.
// B (32x16, KxN): lane L holds col N=L&15 with the same K split — so if the
// B tile is stored TRANSPOSED in LDS ([N][K], N-major) the per-lane access
// pattern is identical to A's: two contiguous 16-byte runs -> ds_load_b128.
// ---------------------------------------------------------------------------
__device__ __forceinline__ v16h fragRowMajor(const _Float16* p, int ld) {
    int lane = threadIdx.x & 31;
    const _Float16* row = p + (size_t)(lane & 15) * ld + ((lane >> 4) << 3);
    v16h a;
#pragma unroll
    for (int i = 0; i < 8; ++i) { a[i] = row[i]; a[i + 8] = row[16 + i]; }
    return a;
}

// float atomic max via sign-split integer atomics (works for any init/ordering)
__device__ __forceinline__ void atomicMaxF(float* addr, float val) {
    if (val >= 0.0f) atomicMax((int*)addr, __float_as_int(val));
    else             atomicMin((unsigned int*)addr, __float_as_uint(val));
}

// ---------------------------------------------------------------------------
// Generic mixed-precision GEMM: C[M,N] = act(A[M,K] @ B[K,N] + bias)
// block = 128 threads (4 waves); block tile 64x64; K step 32.
// A tile row-major [M][K]; B tile stored transposed [N][K] for vector loads.
// ---------------------------------------------------------------------------
#define TM 64
#define TN 64
#define TK 32
__launch_bounds__(128)
__global__ void k_gemm(const float* __restrict__ A, int lda,
                       const float* __restrict__ Bm, int ldb,
                       const float* __restrict__ bias,
                       float* __restrict__ C, int ldc,
                       int M, int Nn, int Kd, int act) {
    __shared__ _Float16 sA[TM][TK + 8];
    __shared__ _Float16 sBt[TN][TK + 8];   // transposed: [n][k]
    const int bm = blockIdx.y * TM, bn = blockIdx.x * TN;
    const int tid = threadIdx.x, wave = tid >> 5, lane = tid & 31;

    v8f acc[4] = {};

    for (int k0 = 0; k0 < Kd; k0 += TK) {
        // prefetch next A tile (speculative, lowers to global_prefetch_b8)
        int pr = bm + (tid & 63);
        if (k0 + TK < Kd && pr < M)
            __builtin_prefetch(&A[(size_t)pr * lda + k0 + TK], 0, 0);
#pragma unroll 4
        for (int i = tid; i < TM * TK; i += 128) {
            int r = i >> 5, c = i & 31;      // TK == 32
            float v = (bm + r < M && k0 + c < Kd) ? A[(size_t)(bm + r) * lda + k0 + c] : 0.f;
            sA[r][c] = (_Float16)v;
        }
#pragma unroll 4
        for (int i = tid; i < TK * TN; i += 128) {
            int r = i / TN, c = i % TN;      // global-coalesced read order
            float v = (k0 + r < Kd && bn + c < Nn) ? Bm[(size_t)(k0 + r) * ldb + bn + c] : 0.f;
            sBt[c][r] = (_Float16)v;         // scatter into transposed tile
        }
        __syncthreads();

        v16h af = fragRowMajor(&sA[wave * 16][0], TK + 8);
#pragma unroll
        for (int nt = 0; nt < 4; ++nt) {
            v16h bf = fragRowMajor(&sBt[nt * 16][0], TK + 8);
            acc[nt] = __builtin_amdgcn_wmma_f32_16x16x32_f16(
                false, af, false, bf, (short)0, acc[nt], false, false);
        }
        __syncthreads();
    }

    const int rbase = bm + wave * 16 + ((lane >> 4) << 3);
#pragma unroll
    for (int nt = 0; nt < 4; ++nt) {
        int col = bn + nt * 16 + (lane & 15);
        if (col >= Nn) continue;
        float bv = bias ? bias[col] : 0.f;
#pragma unroll
        for (int r = 0; r < 8; ++r) {
            int row = rbase + r;
            if (row < M) {
                float v = acc[nt][r] + bv;
                if (act == 1) v = fmaxf(v, 0.f);
                C[(size_t)row * ldc + col] = v;
            }
        }
    }
}

// ---------------------------------------------------------------------------
// Conv1d (NCH, OIH, valid) + ReLU + max-over-time, via WMMA.
// grid (8, B): block handles 128 positions x 32 filters for one batch.
// A[t, k*128+c] = emb[target[b, t+k], c];  B[k*128+c, f] = Wc[f, c, k]
// Weight tile stored transposed in LDS ([f][k]) for vectorized frag loads.
// ---------------------------------------------------------------------------
__launch_bounds__(128)
__global__ void k_conv(const int* __restrict__ target,
                       const float* __restrict__ emb,
                       const float* __restrict__ Wc,
                       const float* __restrict__ bc,
                       float* __restrict__ xt0 /* [B,32], pre-zeroed */) {
    __shared__ _Float16 sX[135][136];
    __shared__ _Float16 sWt[32][40];       // transposed: [filter][k-chunk]
    __shared__ float smax[128][33];
    __shared__ float pmax[4][32];

    const int b = blockIdx.y;
    const int p0 = blockIdx.x * 128;
    const int tid = threadIdx.x, wave = tid >> 5, lane = tid & 31;

    // gather embedding window [p0, p0+135) x 128 channels into LDS (f16)
    for (int i = tid; i < 135 * 128; i += 128) {
        int t = i >> 7, c = i & 127;
        int pos = p0 + t;
        float v = (pos < SEQ) ? emb[(size_t)target[(size_t)b * SEQ + pos] * EMB + c] : 0.f;
        sX[t][c] = (_Float16)v;
    }
    __syncthreads();

    v8f acc[2][2] = {};
    for (int kc = 0; kc < 32; ++kc) {              // K = 1024 = 32 chunks of 32
        int kshift = kc >> 2;                      // filter tap 0..7
        int cbase  = (kc & 3) << 5;                // channel base 0/32/64/96
        for (int i = tid; i < 32 * 32; i += 128) {
            int f = i >> 5, r = i & 31;            // contiguous LDS writes
            sWt[f][r] = (_Float16)Wc[(size_t)f * (EMB * KW) + (size_t)(cbase + r) * KW + kshift];
        }
        __syncthreads();
        v16h b0 = fragRowMajor(&sWt[0][0], 40);
        v16h b1 = fragRowMajor(&sWt[16][0], 40);
#pragma unroll
        for (int mt = 0; mt < 2; ++mt) {
            v16h a = fragRowMajor(&sX[wave * 32 + mt * 16 + kshift][cbase], 136);
            acc[mt][0] = __builtin_amdgcn_wmma_f32_16x16x32_f16(
                false, a, false, b0, (short)0, acc[mt][0], false, false);
            acc[mt][1] = __builtin_amdgcn_wmma_f32_16x16x32_f16(
                false, a, false, b1, (short)0, acc[mt][1], false, false);
        }
        __syncthreads();
    }

    // bias + relu; invalid positions contribute 0 (safe: relu >= 0)
#pragma unroll
    for (int mt = 0; mt < 2; ++mt)
#pragma unroll
        for (int nt = 0; nt < 2; ++nt) {
            int f = nt * 16 + (lane & 15);
#pragma unroll
            for (int r = 0; r < 8; ++r) {
                int rowT = wave * 32 + mt * 16 + r + ((lane >> 4) << 3);
                float v = fmaxf(acc[mt][nt][r] + bc[f], 0.f);
                if (p0 + rowT >= CONV_T) v = 0.f;
                smax[rowT][f] = v;
            }
        }
    __syncthreads();

    int f = tid & 31, seg = tid >> 5;
    float m = 0.f;
    for (int i = 0; i < 32; ++i) m = fmaxf(m, smax[seg * 32 + i][f]);
    pmax[seg][f] = m;
    __syncthreads();
    if (tid < 32) {
        float mm = fmaxf(fmaxf(pmax[0][tid], pmax[1][tid]),
                         fmaxf(pmax[2][tid], pmax[3][tid]));
        atomicMax((int*)&xt0[(size_t)b * NF + tid], __float_as_int(mm)); // values >= 0
    }
}

// ---------------------------------------------------------------------------
// GAT helper kernels
// ---------------------------------------------------------------------------
__global__ void k_fill(float* __restrict__ p, float v, size_t n) {
    size_t i = (size_t)blockIdx.x * blockDim.x + threadIdx.x;
    if (i < n) p[i] = v;
}

__global__ void k_att_logits(const float* __restrict__ h,
                             const float* __restrict__ a_s,
                             const float* __restrict__ a_d,
                             float* __restrict__ as_o, float* __restrict__ ad_o,
                             int n, int H, int C) {
    int idx = blockIdx.x * blockDim.x + threadIdx.x;
    if (idx >= n * H) return;
    int node = idx / H, hh = idx - node * H;
    const float* hp  = h + (size_t)node * H * C + (size_t)hh * C;
    const float* asp = a_s + hh * C;
    const float* adp = a_d + hh * C;
    float s = 0.f, d = 0.f;
    for (int c = 0; c < C; ++c) { float v = hp[c]; s += v * asp[c]; d += v * adp[c]; }
    as_o[idx] = s; ad_o[idx] = d;
}

__device__ __forceinline__ void edge_sd(const int* ei, int e, int& s, int& d) {
    if (e < E_EDGES) { s = ei[e]; d = ei[E_EDGES + e]; }
    else             { s = e - E_EDGES; d = s; }
}

__global__ void k_edge_max(const int* __restrict__ ei,
                           const float* __restrict__ as_, const float* __restrict__ ad_,
                           float* __restrict__ emax, int H) {
    int idx = blockIdx.x * blockDim.x + threadIdx.x;
    if (idx >= E_TOT * H) return;
    int e = idx / H, h = idx - e * H;
    int s, d; edge_sd(ei, e, s, d);
    float v = as_[s * H + h] + ad_[d * H + h];
    v = v > 0.f ? v : SLOPE * v;
    atomicMaxF(&emax[d * H + h], v);
}

__global__ void k_edge_sum(const int* __restrict__ ei,
                           const float* __restrict__ as_, const float* __restrict__ ad_,
                           const float* __restrict__ emax,
                           float* __restrict__ exbuf, float* __restrict__ esum, int H) {
    int idx = blockIdx.x * blockDim.x + threadIdx.x;
    if (idx >= E_TOT * H) return;
    int e = idx / H, h = idx - e * H;
    int s, d; edge_sd(ei, e, s, d);
    float v = as_[s * H + h] + ad_[d * H + h];
    v = v > 0.f ? v : SLOPE * v;
    float ex = expf(v - emax[d * H + h]);
    exbuf[idx] = ex;
    atomicAdd(&esum[d * H + h], ex);
}

__global__ void k_edge_scatter(const int* __restrict__ ei,
                               const float* __restrict__ hsrc,
                               const float* __restrict__ exbuf,
                               const float* __restrict__ esum,
                               float* __restrict__ outacc, int H, int C) {
    int e = blockIdx.x;
    int s, d; edge_sd(ei, e, s, d);
    int HC = H * C;
    for (int j = threadIdx.x; j < HC; j += blockDim.x) {
        int h = j / C;
        float alpha = exbuf[(size_t)e * H + h] / (esum[d * H + h] + 1e-16f);
        atomicAdd(&outacc[(size_t)d * HC + j], alpha * hsrc[(size_t)s * HC + j]);
    }
}

// act: 1 = relu, 2 = elu
__global__ void k_bias_act(float* __restrict__ x, const float* __restrict__ b,
                           size_t rows, int cols, int act) {
    size_t idx = (size_t)blockIdx.x * blockDim.x + threadIdx.x;
    if (idx >= rows * (size_t)cols) return;
    int c = (int)(idx % cols);
    float v = x[idx] + b[c];
    if (act == 1)      v = fmaxf(v, 0.f);
    else if (act == 2) v = v > 0.f ? v : expf(v) - 1.f;
    x[idx] = v;
}

__global__ void k_pool_max(const float* __restrict__ h, float* __restrict__ xg) {
    int b = blockIdx.x, c = threadIdx.x;
    float m = -1e30f;
    for (int i = 0; i < NPG; ++i)
        m = fmaxf(m, h[((size_t)(b * NPG + i)) * OUTC + c]);
    xg[(size_t)b * OUTC + c] = m;
}

__global__ void k_pack_hn(const float* __restrict__ xg, const float* __restrict__ xt,
                          float* __restrict__ hn) {
    int idx = blockIdx.x * blockDim.x + threadIdx.x;
    if (idx >= B_GR * OUTC) return;
    int b = idx >> 7, c = idx & 127;
    hn[(size_t)b * 256 + c]       = xg[idx];
    hn[(size_t)b * 256 + 128 + c] = xt[idx];
}

// cross-attention over the 2 tokens of each pair (4 heads, dh=32)
__launch_bounds__(256)
__global__ void k_attn(const float* __restrict__ q, const float* __restrict__ k,
                       const float* __restrict__ v, float* __restrict__ o) {
    __shared__ float sq[2][128], sk[2][128], sv[2][128];
    int b = blockIdx.x, tid = threadIdx.x;
    int tok = tid >> 7, c = tid & 127;
    size_t base = (size_t)(b * 2 + tok) * 128 + c;
    sq[tok][c] = q[base]; sk[tok][c] = k[base]; sv[tok][c] = v[base];
    __syncthreads();
    int h = c >> 5;
    float s0 = 0.f, s1 = 0.f;
    const float* qp = &sq[tok][h * 32];
    const float* k0 = &sk[0][h * 32];
    const float* k1 = &sk[1][h * 32];
#pragma unroll
    for (int d = 0; d < 32; ++d) { float qq = qp[d]; s0 += qq * k0[d]; s1 += qq * k1[d]; }
    const float scale = 0.17677669529663687f;   // 1/sqrt(32)
    s0 *= scale; s1 *= scale;
    float m = fmaxf(s0, s1);
    float e0 = expf(s0 - m), e1 = expf(s1 - m);
    o[base] = (e0 * sv[0][c] + e1 * sv[1][c]) / (e0 + e1);
}

// residual + layernorm, write straight into concat layout xc[B][256]
__launch_bounds__(128)
__global__ void k_ln_res(const float* __restrict__ hn, const float* __restrict__ ao,
                         const float* __restrict__ g, const float* __restrict__ bb,
                         float* __restrict__ xc) {
    __shared__ float red[128];
    int row = blockIdx.x, c = threadIdx.x;
    float t = hn[(size_t)row * 128 + c] + ao[(size_t)row * 128 + c];
    red[c] = t; __syncthreads();
    for (int s = 64; s > 0; s >>= 1) { if (c < s) red[c] += red[c + s]; __syncthreads(); }
    float mu = red[0] * (1.f / 128.f);
    __syncthreads();
    float dm = t - mu;
    red[c] = dm * dm; __syncthreads();
    for (int s = 64; s > 0; s >>= 1) { if (c < s) red[c] += red[c + s]; __syncthreads(); }
    float var = red[0] * (1.f / 128.f);
    float y = dm * rsqrtf(var + 1e-5f) * g[c] + bb[c];
    int b = row >> 1, tok = row & 1;
    xc[(size_t)b * 256 + tok * 128 + c] = y;
}

__launch_bounds__(256)
__global__ void k_final(const float* __restrict__ f2, const float* __restrict__ Wout,
                        const float* __restrict__ bout, float* __restrict__ out) {
    __shared__ float red[256];
    int b = blockIdx.x, tid = threadIdx.x;
    red[tid] = f2[(size_t)b * 256 + tid] * Wout[tid];
    __syncthreads();
    for (int s = 128; s > 0; s >>= 1) { if (tid < s) red[tid] += red[tid + s]; __syncthreads(); }
    if (tid == 0) out[b] = red[0] + bout[0];
}

// ---------------------------------------------------------------------------
// Host orchestration
// ---------------------------------------------------------------------------
extern "C" void kernel_launch(void* const* d_in, const int* in_sizes, int n_in,
                              void* d_out, int out_size, void* d_ws, size_t ws_size,
                              hipStream_t stream) {
    const float* x      = (const float*)d_in[0];
    const int*   ei     = (const int*)d_in[1];
    /* batch (d_in[2]) unused: graphs are contiguous NPG blocks */
    const int*   target = (const int*)d_in[3];
    const float* W1 = (const float*)d_in[4];
    const float* a_src1 = (const float*)d_in[5];
    const float* a_dst1 = (const float*)d_in[6];
    const float* b1 = (const float*)d_in[7];
    const float* W2 = (const float*)d_in[8];
    const float* a_src2 = (const float*)d_in[9];
    const float* a_dst2 = (const float*)d_in[10];
    const float* b2 = (const float*)d_in[11];
    const float* Wg = (const float*)d_in[12];
    const float* bg = (const float*)d_in[13];
    const float* emb = (const float*)d_in[14];
    const float* Wc = (const float*)d_in[15];
    const float* bc = (const float*)d_in[16];
    const float* Wxt = (const float*)d_in[17];
    const float* bxt = (const float*)d_in[18];
    const float* Wq = (const float*)d_in[19];
    const float* bq = (const float*)d_in[20];
    const float* Wk = (const float*)d_in[21];
    const float* bk = (const float*)d_in[22];
    const float* Wv = (const float*)d_in[23];
    const float* bv = (const float*)d_in[24];
    const float* Wo = (const float*)d_in[25];
    const float* bo = (const float*)d_in[26];
    const float* ln_g = (const float*)d_in[27];
    const float* ln_b = (const float*)d_in[28];
    const float* Wf1 = (const float*)d_in[29];
    const float* bf1 = (const float*)d_in[30];
    const float* Wf2 = (const float*)d_in[31];
    const float* bf2 = (const float*)d_in[32];
    const float* Wout = (const float*)d_in[33];
    const float* bout = (const float*)d_in[34];
    float* out = (float*)d_out;

    char* ws = (char*)d_ws;
    size_t off = 0;
    auto alloc_f = [&](size_t n) -> float* {
        off = (off + 255) & ~(size_t)255;
        float* p = (float*)(ws + off);
        off += n * sizeof(float);
        return p;
    };

    float* h1    = alloc_f((size_t)N_NODES * H1C);
    float* acc1  = alloc_f((size_t)N_NODES * H1C);
    float* as1   = alloc_f((size_t)N_NODES * HEADS);
    float* ad1   = alloc_f((size_t)N_NODES * HEADS);
    float* emax1 = alloc_f((size_t)N_NODES * HEADS);
    float* esum1 = alloc_f((size_t)N_NODES * HEADS);
    float* exbuf = alloc_f((size_t)E_TOT * HEADS);   // reused for layer 2 (E_TOT*1)
    float* h2    = alloc_f((size_t)N_NODES * OUTC);
    float* acc2  = alloc_f((size_t)N_NODES * OUTC);
    float* as2   = alloc_f(N_NODES);
    float* ad2   = alloc_f(N_NODES);
    float* emax2 = alloc_f(N_NODES);
    float* esum2 = alloc_f(N_NODES);
    float* xgp   = alloc_f((size_t)B_GR * OUTC);
    float* xg    = alloc_f((size_t)B_GR * OUTC);
    float* xt0   = alloc_f((size_t)B_GR * NF);
    float* xtb   = alloc_f((size_t)B_GR * OUTC);
    float* hn    = alloc_f((size_t)B_GR * 2 * OUTC);
    float* qb    = alloc_f((size_t)B_GR * 2 * OUTC);
    float* kb    = alloc_f((size_t)B_GR * 2 * OUTC);
    float* vb    = alloc_f((size_t)B_GR * 2 * OUTC);
    float* ob    = alloc_f((size_t)B_GR * 2 * OUTC);
    float* aob   = alloc_f((size_t)B_GR * 2 * OUTC);
    float* xc    = alloc_f((size_t)B_GR * 256);
    float* f1b   = alloc_f((size_t)B_GR * 1024);
    float* f2b   = alloc_f((size_t)B_GR * 256);

    auto gemm = [&](const float* A, int lda, const float* Bm, int ldb,
                    const float* bias, float* C, int ldc,
                    int M, int Nn, int Kd, int act) {
        dim3 g((Nn + TN - 1) / TN, (M + TM - 1) / TM);
        k_gemm<<<g, 128, 0, stream>>>(A, lda, Bm, ldb, bias, C, ldc, M, Nn, Kd, act);
    };
    auto fill = [&](float* p, float v, size_t n) {
        k_fill<<<(unsigned)((n + 255) / 256), 256, 0, stream>>>(p, v, n);
    };

    // ---- GAT layer 1 (10 heads, 78 channels) ----
    gemm(x, FXD, W1, H1C, nullptr, h1, H1C, N_NODES, H1C, FXD, 0);
    {
        int tot = N_NODES * HEADS;
        k_att_logits<<<(tot + 255) / 256, 256, 0, stream>>>(h1, a_src1, a_dst1, as1, ad1,
                                                            N_NODES, HEADS, FXD);
    }
    fill(emax1, -1e30f, (size_t)N_NODES * HEADS);
    fill(esum1, 0.f, (size_t)N_NODES * HEADS);
    fill(acc1, 0.f, (size_t)N_NODES * H1C);
    {
        int tot = E_TOT * HEADS;
        k_edge_max<<<(tot + 255) / 256, 256, 0, stream>>>(ei, as1, ad1, emax1, HEADS);
        k_edge_sum<<<(tot + 255) / 256, 256, 0, stream>>>(ei, as1, ad1, emax1, exbuf, esum1, HEADS);
        k_edge_scatter<<<E_TOT, 256, 0, stream>>>(ei, h1, exbuf, esum1, acc1, HEADS, FXD);
    }
    {
        size_t tot = (size_t)N_NODES * H1C;
        k_bias_act<<<(unsigned)((tot + 255) / 256), 256, 0, stream>>>(acc1, b1, N_NODES, H1C, 2);
    }

    // ---- GAT layer 2 (1 head, 128 channels) ----
    gemm(acc1, H1C, W2, OUTC, nullptr, h2, OUTC, N_NODES, OUTC, H1C, 0);
    k_att_logits<<<(N_NODES + 255) / 256, 256, 0, stream>>>(h2, a_src2, a_dst2, as2, ad2,
                                                            N_NODES, 1, OUTC);
    fill(emax2, -1e30f, N_NODES);
    fill(esum2, 0.f, N_NODES);
    fill(acc2, 0.f, (size_t)N_NODES * OUTC);
    k_edge_max<<<(E_TOT + 255) / 256, 256, 0, stream>>>(ei, as2, ad2, emax2, 1);
    k_edge_sum<<<(E_TOT + 255) / 256, 256, 0, stream>>>(ei, as2, ad2, emax2, exbuf, esum2, 1);
    k_edge_scatter<<<E_TOT, 128, 0, stream>>>(ei, h2, exbuf, esum2, acc2, 1, OUTC);
    {
        size_t tot = (size_t)N_NODES * OUTC;
        k_bias_act<<<(unsigned)((tot + 255) / 256), 256, 0, stream>>>(acc2, b2, N_NODES, OUTC, 1);
    }

    // ---- pool + ligand projection ----
    k_pool_max<<<B_GR, OUTC, 0, stream>>>(acc2, xgp);
    gemm(xgp, OUTC, Wg, OUTC, bg, xg, OUTC, B_GR, OUTC, OUTC, 1);

    // ---- protein branch ----
    fill(xt0, 0.f, (size_t)B_GR * NF);
    k_conv<<<dim3(8, B_GR), 128, 0, stream>>>(target, emb, Wc, bc, xt0);
    gemm(xt0, NF, Wxt, OUTC, bxt, xtb, OUTC, B_GR, OUTC, NF, 0);

    // ---- cross-attention fusion ----
    k_pack_hn<<<(B_GR * OUTC + 255) / 256, 256, 0, stream>>>(xg, xtb, hn);
    gemm(hn, OUTC, Wq, OUTC, bq, qb, OUTC, 2 * B_GR, OUTC, OUTC, 0);
    gemm(hn, OUTC, Wk, OUTC, bk, kb, OUTC, 2 * B_GR, OUTC, OUTC, 0);
    gemm(hn, OUTC, Wv, OUTC, bv, vb, OUTC, 2 * B_GR, OUTC, OUTC, 0);
    k_attn<<<B_GR, 256, 0, stream>>>(qb, kb, vb, ob);
    gemm(ob, OUTC, Wo, OUTC, bo, aob, OUTC, 2 * B_GR, OUTC, OUTC, 0);
    k_ln_res<<<2 * B_GR, 128, 0, stream>>>(hn, aob, ln_g, ln_b, xc);

    // ---- MLP head ----
    gemm(xc, 256, Wf1, 1024, bf1, f1b, 1024, B_GR, 1024, 256, 1);
    gemm(f1b, 1024, Wf2, 256, bf2, f2b, 256, B_GR, 256, 1024, 1);
    k_final<<<B_GR, 256, 0, stream>>>(f2b, Wout, bout, out);
}
